// PredictionNetworkForImageInput_7541962572369
// MI455X (gfx1250) — matrix operand order repaired and checked by reading, not compiled, not run on hardware
//
#include <hip/hip_runtime.h>
#include <hip/hip_bf16.h>

typedef __attribute__((ext_vector_type(16))) _Float16 v16h;
typedef __attribute__((ext_vector_type(8)))  _Float16 h8;
typedef __attribute__((ext_vector_type(8)))  float    v8f;
typedef __attribute__((ext_vector_type(4)))  float    f4;

// ---------------- problem constants ----------------
constexpr int NB    = 8;     // meta-batch (tasks)
constexpr int NT    = 5;     // inner train steps
constexpr int NL    = 16;    // test examples per task
constexpr int NCLSK = 100;   // classes
constexpr int NCLSP = 128;   // classes padded for WMMA N-tiling / unconditional stores
constexpr int FEATK = 4096;  // 256*4*4

// =====================================================================
// Elementwise / small kernels
// =====================================================================
__global__ void fill_zero(float* __restrict__ p, int n) {
  int i = blockIdx.x * blockDim.x + threadIdx.x;
  if (i < n) p[i] = 0.f;
}

__global__ void bcast_param(const float* __restrict__ src, float* __restrict__ dst, int n) {
  int i = blockIdx.x * blockDim.x + threadIdx.x;
  if (i < n * NB) dst[i] = src[i % n];
}

// broadcast weights with K padded to Kp (zero fill); also used to pad 1-row vectors
__global__ void bcast_w_pad(const float* __restrict__ src, float* __restrict__ dst,
                            int rows, int K, int Kp) {
  int i = blockIdx.x * blockDim.x + threadIdx.x;
  int per = rows * Kp;
  if (i < per * NB) {
    int rk = i % per;
    int r = rk / Kp, k = rk - r * Kp;
    dst[i] = (k < K) ? src[r * K + k] : 0.f;
  }
}

// broadcast with row count padded to rowsP (zero fill)
__global__ void bcast_rows_pad(const float* __restrict__ src, float* __restrict__ dst,
                               int rows, int rowsP, int rowlen) {
  int i = blockIdx.x * blockDim.x + threadIdx.x;
  int per = rowsP * rowlen;
  if (i < per * NB) {
    int rk = i % per;
    int r = rk / rowlen, c = rk - r * rowlen;
    dst[i] = (r < rows) ? src[r * rowlen + c] : 0.f;
  }
}

__global__ void gather_copy(const float* __restrict__ src, float* __restrict__ dst,
                            long base, long imgStride, int perImg, int nimg) {
  int i = blockIdx.x * blockDim.x + threadIdx.x;
  if (i < perImg * nimg) {
    int img = i / perImg;
    int k   = i - img * perImg;
    dst[i] = src[base + (long)img * imgStride + k];
  }
}

__global__ void gate_mul(const float* __restrict__ a, float* __restrict__ f,
                         const float* __restrict__ gate, long gateBase, long gateImgStride,
                         long dstImgStride, int perImg, int nimg) {
  int i = blockIdx.x * blockDim.x + threadIdx.x;
  if (i < perImg * nimg) {
    int img = i / perImg;
    int m   = i - img * perImg;
    f[(long)img * dstImgStride + m] = a[i] * gate[gateBase + (long)img * gateImgStride + m];
  }
}

__global__ void relu_mask(float* __restrict__ g, const float* __restrict__ a, int n) {
  int i = blockIdx.x * blockDim.x + threadIdx.x;
  if (i < n) g[i] = (a[i] > 0.f) ? g[i] : 0.f;
}

__global__ void write_lr(const float* __restrict__ log_lr, float* __restrict__ dst) {
  if (threadIdx.x == 0 && blockIdx.x == 0) dst[0] = __expf(log_lr[0]);
}

// =====================================================================
// Implicit-GEMM conv forward: y = relu(conv(x, W) + b)
// One wave per 16(OC) x 16*NTILE(pixels) tile; K = IC*9 padded to Kp.
// Weights pre-padded (zero) to Kp, so A staging is unconditional b128.
// B tile stored n-major so fragment builds are contiguous ds_load_b128.
// =====================================================================
template <int NTILE>
__global__ __launch_bounds__(32)
void conv_fwd_wmma(const float* __restrict__ x, const float* __restrict__ fwW,
                   const float* __restrict__ fwB, float* __restrict__ y,
                   int ICc, int OCc, int Hin, int Win, int Hout, int Wout,
                   int woShift, int stride, int K, int Kp,
                   int taskMul, int taskBase, long wTaskStride, long bTaskStride) {
  __shared__ _Float16 As[16][32];              // [oc][k]
  __shared__ _Float16 BsT[16 * NTILE][32];     // [pixel][k]  (n-major!)

  const int lane = threadIdx.x;
  const int img  = blockIdx.z;
  const int task = taskBase + img * taskMul;
  const int oc0  = blockIdx.y * 16;
  const int p0   = blockIdx.x * (16 * NTILE);
  const int HWin = Hin * Win;
  const int Npix = Hout * Wout;

  const float* W    = fwW + (long)task * wTaskStride;
  const float* bias = fwB + (long)task * bTaskStride;
  const float* xi   = x + (long)img * ICc * HWin;

  v8f acc[NTILE];
  #pragma unroll
  for (int nt = 0; nt < NTILE; ++nt) acc[nt] = {};

  const int half = lane >> 4;   // 0 | 1
  const int l16  = lane & 15;

  // B staging assignment:
  //  NTILE==2: lane stages pixel n=lane, k = 0..31
  //  NTILE==1: lane stages pixel n=lane&15, k = half*16 .. half*16+15
  const int nloc = (NTILE == 2) ? lane : l16;
  const int kbeg = (NTILE == 2) ? 0 : half * 16;
  const int kcnt = (NTILE == 2) ? 32 : 16;
  const int pB   = p0 + nloc;
  const int hoB  = pB >> woShift;
  const int woB  = pB - (hoB << woShift);

  // A staging assignment: lane stages row m=lane/2, k-halves kb=(lane&1)*16
  const int mA  = lane >> 1;
  const int kbA = (lane & 1) * 16;
  const float* wrowA = W + (long)(oc0 + mA) * Kp + kbA;

  for (int k0 = 0; k0 < K; k0 += 32) {
    // ---- stage A (unconditional, vector loads) ----
    {
      const f4* wv = (const f4*)(wrowA + k0);
      f4 w0 = wv[0], w1 = wv[1], w2 = wv[2], w3 = wv[3];
      h8 h0, h1;
      #pragma unroll
      for (int e = 0; e < 4; ++e) {
        h0[e]     = (_Float16)w0[e];
        h0[4 + e] = (_Float16)w1[e];
        h1[e]     = (_Float16)w2[e];
        h1[4 + e] = (_Float16)w3[e];
      }
      *(h8*)&As[mA][kbA]     = h0;
      *(h8*)&As[mA][kbA + 8] = h1;
      if (k0 + 32 < K) __builtin_prefetch(wrowA + k0 + 32, 0, 1);
    }
    // ---- stage B (im2col, clamped unconditional loads, h8 stores) ----
    #pragma unroll
    for (int c = 0; c < kcnt / 8; ++c) {
      h8 hv;
      #pragma unroll
      for (int e = 0; e < 8; ++e) {
        int k    = k0 + kbeg + c * 8 + e;
        int kval = (k < K);
        int kc   = kval ? k : 0;
        int ic = kc / 9, rs = kc - ic * 9;
        int r = rs / 3, s = rs - r * 3;
        int hi = hoB * stride - 1 + r;
        int wi = woB * stride - 1 + s;
        int inb = kval & (hi >= 0) & (hi < Hin) & (wi >= 0) & (wi < Win);
        int hc = hi < 0 ? 0 : (hi >= Hin ? Hin - 1 : hi);
        int wc = wi < 0 ? 0 : (wi >= Win ? Win - 1 : wi);
        float v = xi[(long)ic * HWin + hc * Win + wc];
        hv[e] = (_Float16)(inb ? v : 0.f);
      }
      *(h8*)&BsT[nloc][kbeg + c * 8] = hv;
    }
    __syncthreads();

    // ---- fragments (ISA 7.12.2 layouts) + WMMA ----
    h8 alo = *(const h8*)&As[l16][half * 8];
    h8 ahi = *(const h8*)&As[l16][16 + half * 8];
    v16h a = __builtin_shufflevector(alo, ahi, 0, 1, 2, 3, 4, 5, 6, 7,
                                     8, 9, 10, 11, 12, 13, 14, 15);
    #pragma unroll
    for (int nt = 0; nt < NTILE; ++nt) {
      h8 b0 = *(const h8*)&BsT[nt * 16 + l16][half * 8];
      h8 b1 = *(const h8*)&BsT[nt * 16 + l16][16 + half * 8];
      v16h b = __builtin_shufflevector(b0, b1, 0, 1, 2, 3, 4, 5, 6, 7,
                                       8, 9, 10, 11, 12, 13, 14, 15);
      acc[nt] = __builtin_amdgcn_wmma_f32_16x16x32_f16(false, a, false, b,
                                                       (short)0, acc[nt], false, false);
    }
    __syncthreads();
  }

  // ---- epilogue: bias + ReLU, store (D: VGPR r -> M=half*8+r, N=lane&15) ----
  #pragma unroll
  for (int nt = 0; nt < NTILE; ++nt) {
    #pragma unroll
    for (int r = 0; r < 8; ++r) {
      int m = half * 8 + r;
      float v = acc[nt][r] + bias[oc0 + m];
      v = v > 0.f ? v : 0.f;
      y[((long)img * OCc + (oc0 + m)) * Npix + (p0 + nt * 16 + l16)] = v;
    }
  }
}

// =====================================================================
// Head GEMM: logits[row][n] = sum_m feat[row][m] * Wout[n][m] + bout[n]
// M = 16 rows (zero-padded), N = 128 padded classes (2 tiles), K = 4096.
// Bias padded to 128, logits buffer padded 16x128 -> fully unconditional.
// =====================================================================
__global__ __launch_bounds__(32)
void head_fwd_wmma(const float* __restrict__ feat, const float* __restrict__ Wo,
                   const float* __restrict__ bo, float* __restrict__ logits,
                   int taskMul, int taskBase,
                   long featBlkStride, long logitBlkStride) {
  __shared__ _Float16 As[16][32];    // [row][k]
  __shared__ _Float16 BsT[32][32];   // [class][k]

  const int lane = threadIdx.x;
  const int task = taskBase + (int)blockIdx.y * taskMul;
  const int n0   = blockIdx.x * 32;
  const float* W  = Wo + (long)task * NCLSP * FEATK;
  const float* bb = bo + (long)task * NCLSP;
  const float* fp = feat + (long)blockIdx.y * featBlkStride;
  float* lp       = logits + (long)blockIdx.y * logitBlkStride;

  v8f acc[2] = {{}, {}};
  const int half = lane >> 4;
  const int l16  = lane & 15;

  const int mA  = lane >> 1;
  const int kbA = (lane & 1) * 16;
  const float* frow = fp + (long)mA * FEATK + kbA;
  const float* wrow = W + (long)(n0 + lane) * FEATK;

  for (int k0 = 0; k0 < FEATK; k0 += 32) {
    // ---- stage A: feature rows (pre-zero-padded to 16 rows) ----
    {
      const f4* fv = (const f4*)(frow + k0);
      f4 w0 = fv[0], w1 = fv[1], w2 = fv[2], w3 = fv[3];
      h8 h0, h1;
      #pragma unroll
      for (int e = 0; e < 4; ++e) {
        h0[e]     = (_Float16)w0[e];
        h0[4 + e] = (_Float16)w1[e];
        h1[e]     = (_Float16)w2[e];
        h1[4 + e] = (_Float16)w3[e];
      }
      *(h8*)&As[mA][kbA]     = h0;
      *(h8*)&As[mA][kbA + 8] = h1;
    }
    // ---- stage B: one class row per lane (rows pre-padded to 128) ----
    #pragma unroll
    for (int c = 0; c < 4; ++c) {
      const f4* wv = (const f4*)(wrow + k0 + c * 8);
      f4 wa = wv[0], wb = wv[1];
      h8 hv;
      #pragma unroll
      for (int e = 0; e < 4; ++e) { hv[e] = (_Float16)wa[e]; hv[4 + e] = (_Float16)wb[e]; }
      *(h8*)&BsT[lane][c * 8] = hv;
    }
    __syncthreads();

    h8 alo = *(const h8*)&As[l16][half * 8];
    h8 ahi = *(const h8*)&As[l16][16 + half * 8];
    v16h a = __builtin_shufflevector(alo, ahi, 0, 1, 2, 3, 4, 5, 6, 7,
                                     8, 9, 10, 11, 12, 13, 14, 15);
    #pragma unroll
    for (int nt = 0; nt < 2; ++nt) {
      h8 b0 = *(const h8*)&BsT[nt * 16 + l16][half * 8];
      h8 b1 = *(const h8*)&BsT[nt * 16 + l16][16 + half * 8];
      v16h b = __builtin_shufflevector(b0, b1, 0, 1, 2, 3, 4, 5, 6, 7,
                                       8, 9, 10, 11, 12, 13, 14, 15);
      acc[nt] = __builtin_amdgcn_wmma_f32_16x16x32_f16(false, a, false, b,
                                                       (short)0, acc[nt], false, false);
    }
    __syncthreads();
  }

  // ---- epilogue: fully unconditional (padded bias + padded logits) ----
  #pragma unroll
  for (int nt = 0; nt < 2; ++nt) {
    int n = n0 + nt * 16 + l16;
    float bv = bb[n];
    #pragma unroll
    for (int r = 0; r < 8; ++r) {
      int m = half * 8 + r;
      lp[(long)m * NCLSP + n] = acc[nt][r] + bv;
    }
  }
}

// =====================================================================
// CE backward (train): dlogit = softmax(logits row 0) - onehot(y)
// logits rows have stride NCLSP
// =====================================================================
__global__ void ce_train(const float* __restrict__ logits, long logitBlkStride,
                         const int* __restrict__ train_y, int t,
                         float* __restrict__ dlogit) {
  const int b = blockIdx.x, tid = threadIdx.x;
  __shared__ float red[128];
  __shared__ float sMax, sSum;
  const float* l = logits + (long)b * logitBlkStride;   // row 0 of task b

  float m = -1e30f;
  for (int n = tid; n < NCLSK; n += 128) m = fmaxf(m, l[n]);
  red[tid] = m; __syncthreads();
  for (int s = 64; s > 0; s >>= 1) { if (tid < s) red[tid] = fmaxf(red[tid], red[tid + s]); __syncthreads(); }
  if (tid == 0) sMax = red[0]; __syncthreads();

  float se = 0.f;
  for (int n = tid; n < NCLSK; n += 128) se += __expf(l[n] - sMax);
  red[tid] = se; __syncthreads();
  for (int s = 64; s > 0; s >>= 1) { if (tid < s) red[tid] += red[tid + s]; __syncthreads(); }
  if (tid == 0) sSum = red[0]; __syncthreads();

  const int y = train_y[b * NT + t];
  for (int n = tid; n < NCLSK; n += 128) {
    float p = __expf(l[n] - sMax) / sSum;
    dlogit[b * NCLSK + n] = p - (n == y ? 1.f : 0.f);
  }
}

// =====================================================================
// CE eval (test): write loss, argmax-correctness and logits into d_out
// d_out: loss[128] @0 | lr @128 | eval[128] @129 | logits[12800] @257
// logits rows have stride NCLSP
// =====================================================================
__global__ void ce_test(const float* __restrict__ logits, const int* __restrict__ test_y,
                        const int* __restrict__ evalflag, int task, float* __restrict__ out) {
  const int l = blockIdx.x, tid = threadIdx.x;
  __shared__ float red[128];
  __shared__ int   ri[128];
  __shared__ float sMax, sSum;
  const float* lg = logits + (long)l * NCLSP;

  float m = -1e30f;
  for (int n = tid; n < NCLSK; n += 128) m = fmaxf(m, lg[n]);
  red[tid] = m; __syncthreads();
  for (int s = 64; s > 0; s >>= 1) { if (tid < s) red[tid] = fmaxf(red[tid], red[tid + s]); __syncthreads(); }
  if (tid == 0) sMax = red[0]; __syncthreads();

  float se = 0.f;
  for (int n = tid; n < NCLSK; n += 128) se += __expf(lg[n] - sMax);
  red[tid] = se; __syncthreads();
  for (int s = 64; s > 0; s >>= 1) { if (tid < s) red[tid] += red[tid + s]; __syncthreads(); }
  if (tid == 0) sSum = red[0]; __syncthreads();

  float bv = -1e30f; int bi = NCLSK;
  for (int n = tid; n < NCLSK; n += 128) { float v = lg[n]; if (v > bv) { bv = v; bi = n; } }
  red[tid] = bv; ri[tid] = bi; __syncthreads();
  for (int s = 64; s > 0; s >>= 1) {
    if (tid < s) {
      if (red[tid + s] > red[tid] || (red[tid + s] == red[tid] && ri[tid + s] < ri[tid])) {
        red[tid] = red[tid + s]; ri[tid] = ri[tid + s];
      }
    }
    __syncthreads();
  }

  const int y = test_y[task * NL + l];
  if (tid == 0) {
    out[task * NL + l]       = -(lg[y] - sMax - __logf(sSum));
    out[129 + task * NL + l] = (evalflag[0] != 0 && ri[0] == y) ? 1.f : 0.f;
  }
  for (int n = tid; n < NCLSK; n += 128)
    out[257 + ((long)(task * NL + l)) * NCLSK + n] = lg[n];
}

// =====================================================================
// Head backward: dact5[m] = gate[m] * sum_n dlogit[n] * Wout[n][m]
// =====================================================================
__global__ void head_bwd_df(const float* __restrict__ dlogit, const float* __restrict__ Wo,
                            const float* __restrict__ gate, long gateBase, long gateStride,
                            float* __restrict__ dact5) {
  const int b = blockIdx.y;
  const int m = blockIdx.x * blockDim.x + threadIdx.x;
  const float* W  = Wo + (long)b * NCLSP * FEATK;
  const float* dl = dlogit + b * NCLSK;
  float acc = 0.f;
  for (int n = 0; n < NCLSK; ++n) acc += dl[n] * W[(long)n * FEATK + m];
  dact5[(long)b * FEATK + m] = acc * gate[gateBase + (long)b * gateStride + m];
}

__global__ void head_upd(float* __restrict__ Wo, float* __restrict__ bo,
                         const float* __restrict__ dlogit, const float* __restrict__ feat,
                         long featBlkStride, const float* __restrict__ log_lr) {
  const int b = blockIdx.y;
  const long i = (long)blockIdx.x * blockDim.x + threadIdx.x;
  const float lr = __expf(log_lr[0]);
  if (i < (long)NCLSK * FEATK) {
    int n = (int)(i / FEATK), m = (int)(i - (long)n * FEATK);
    Wo[(long)b * NCLSP * FEATK + i] -= lr * dlogit[b * NCLSK + n] * feat[(long)b * featBlkStride + m];
    if (m == 0) bo[b * NCLSP + n] -= lr * dlogit[b * NCLSK + n];
  }
}

// =====================================================================
// Conv backward data: dx = conv_transpose(dy, W)  (pad=1, K-padded weights)
// =====================================================================
__global__ void conv_bwd_x(const float* __restrict__ dy, const float* __restrict__ fwW,
                           float* __restrict__ dx,
                           int ICc, int OCc, int Hin, int Win, int Hout, int Wout,
                           int stride, int Kp, long wTaskStride) {
  const int img = blockIdx.y;  // == task in the inner loop
  const int idx = blockIdx.x * blockDim.x + threadIdx.x;
  const int HWin = Hin * Win, Npix = Hout * Wout;
  if (idx >= ICc * HWin) return;
  const int ic = idx / HWin;
  const int hw = idx - ic * HWin;
  const int hi = hw / Win, wi = hw - (hw / Win) * Win;
  const float* W = fwW + (long)img * wTaskStride;
  const float* g = dy + (long)img * OCc * Npix;
  float acc = 0.f;
  for (int r = 0; r < 3; ++r) {
    int th = hi + 1 - r;
    if (th < 0 || (th % stride)) continue;
    int ho = th / stride; if (ho >= Hout) continue;
    for (int s = 0; s < 3; ++s) {
      int tw = wi + 1 - s;
      if (tw < 0 || (tw % stride)) continue;
      int wo = tw / stride; if (wo >= Wout) continue;
      const int pix = ho * Wout + wo;
      for (int oc = 0; oc < OCc; ++oc)
        acc += W[(long)oc * Kp + ic * 9 + r * 3 + s] * g[(long)oc * Npix + pix];
    }
  }
  dx[(long)img * ICc * HWin + idx] = acc;
}

// =====================================================================
// Conv backward weights + fused SGD update (one train image per task)
// =====================================================================
__global__ void conv_bwd_w_upd(const float* __restrict__ dy, const float* __restrict__ x,
                               float* __restrict__ fwW, float* __restrict__ fwB,
                               const float* __restrict__ log_lr,
                               int ICc, int OCc, int Hin, int Win, int Hout, int Wout,
                               int stride, int Kp, long wTaskStride, long bTaskStride) {
  const int oc = blockIdx.x, task = blockIdx.y, tid = threadIdx.x;
  const int K = ICc * 9, Npix = Hout * Wout, HWin = Hin * Win;
  const float* g  = dy + ((long)task * OCc + oc) * Npix;
  const float* xi = x + (long)task * ICc * HWin;
  const float lr = __expf(log_lr[0]);

  for (int k = tid; k < K; k += 256) {
    int ic = k / 9, rs = k - ic * 9;
    int r = rs / 3, s = rs - r * 3;
    const float* xc = xi + (long)ic * HWin;
    float acc = 0.f;
    for (int p = 0; p < Npix; ++p) {
      int ho = p / Wout, wo = p - ho * Wout;
      int hi = ho * stride - 1 + r, wi = wo * stride - 1 + s;
      if (hi >= 0 && hi < Hin && wi >= 0 && wi < Win) acc += g[p] * xc[hi * Win + wi];
    }
    fwW[(long)task * wTaskStride + (long)oc * Kp + k] -= lr * acc;
  }

  __shared__ float red[256];
  float sb = 0.f;
  for (int p = tid; p < Npix; p += 256) sb += g[p];
  red[tid] = sb; __syncthreads();
  for (int s2 = 128; s2 > 0; s2 >>= 1) { if (tid < s2) red[tid] += red[tid + s2]; __syncthreads(); }
  if (tid == 0) fwB[(long)task * bTaskStride + oc] -= lr * red[0];
}

// =====================================================================
// Host orchestration
// =====================================================================
static const int kIC[5] = {3, 32, 64, 128, 256};
static const int kOC[5] = {32, 64, 128, 256, 256};
static const int kHI[5] = {64, 64, 32, 16, 8};
static const int kHO[5] = {64, 32, 16, 8, 4};
static const int kWS[5] = {6, 5, 4, 3, 2};     // log2(Wout)
static const int kST[5] = {1, 2, 2, 2, 2};

extern "C" void kernel_launch(void* const* d_in, const int* in_sizes, int n_in,
                              void* d_out, int out_size, void* d_ws, size_t ws_size,
                              hipStream_t stream) {
  (void)in_sizes; (void)n_in; (void)out_size; (void)ws_size;

  const float* w_in[5]; const float* b_in[5];
  for (int j = 0; j < 5; ++j) { w_in[j] = (const float*)d_in[2 * j]; b_in[j] = (const float*)d_in[2 * j + 1]; }
  const float* Wo_in   = (const float*)d_in[10];
  const float* bo_in   = (const float*)d_in[11];
  const float* log_lr  = (const float*)d_in[12];
  const float* train_x = (const float*)d_in[13];
  const float* test_x  = (const float*)d_in[14];
  const float* train_g = (const float*)d_in[15];
  const float* test_g  = (const float*)d_in[16];
  const int*   train_y = (const int*)d_in[17];
  const int*   test_y  = (const int*)d_in[18];
  const int*   evalflg = (const int*)d_in[19];
  float* out = (float*)d_out;

  int K[5], Kp[5], WP[5], ASZ[5];
  for (int j = 0; j < 5; ++j) {
    K[j]   = kIC[j] * 9;
    Kp[j]  = ((K[j] + 31) / 32) * 32;
    WP[j]  = kOC[j] * Kp[j];
    ASZ[j] = kOC[j] * kHO[j] * kHO[j];
  }
  const int IMG0 = 3 * 64 * 64;

  // ---- ws layout (floats) ----
  float* ws = (float*)d_ws;
  size_t off = 0;
  auto take = [&](size_t n) { float* p = ws + off; off += n; return p; };
  float* fwW[5]; float* fwB[5];
  for (int j = 0; j < 5; ++j) fwW[j] = take((size_t)NB * WP[j]);
  for (int j = 0; j < 5; ++j) fwB[j] = take((size_t)NB * kOC[j]);
  float* fwWo = take((size_t)NB * NCLSP * FEATK);
  float* fwBo = take((size_t)NB * NCLSP);           // bias padded to 128
  float* actT[6];
  actT[0] = take((size_t)NB * IMG0);
  for (int j = 0; j < 5; ++j) actT[j + 1] = take((size_t)NB * ASZ[j]);
  float* featT   = take((size_t)NB * 16 * FEATK);   // 16 rows/task, rows 1..15 stay zero
  float* logitT  = take((size_t)NB * 16 * NCLSP);   // padded 16x128 per task
  float* dlogitT = take((size_t)NB * NCLSK);
  float* dact5   = take((size_t)NB * FEATK);
  float* g0      = take((size_t)NB * 131072);
  float* g1      = take((size_t)NB * 131072);
  float* tpA     = take((size_t)NL * 131072);
  float* tpB     = take((size_t)NL * 131072);
  float* featE   = take((size_t)NL * FEATK);
  float* logitE  = take((size_t)NL * NCLSP);        // padded 16x128

  // ---- broadcast slow weights into per-task fast weights (padded) ----
  for (int j = 0; j < 5; ++j) {
    bcast_w_pad<<<(NB * WP[j] + 255) / 256, 256, 0, stream>>>(w_in[j], fwW[j], kOC[j], K[j], Kp[j]);
    bcast_param<<<(NB * kOC[j] + 255) / 256, 256, 0, stream>>>(b_in[j], fwB[j], kOC[j]);
  }
  bcast_rows_pad<<<(NB * NCLSP * FEATK + 255) / 256, 256, 0, stream>>>(Wo_in, fwWo, NCLSK, NCLSP, FEATK);
  bcast_w_pad<<<(NB * NCLSP + 255) / 256, 256, 0, stream>>>(bo_in, fwBo, 1, NCLSK, NCLSP);
  fill_zero<<<(NB * 16 * FEATK + 255) / 256, 256, 0, stream>>>(featT, NB * 16 * FEATK);

  // ---- inner loop (T SGD steps, one image per task per step) ----
  for (int t = 0; t < NT; ++t) {
    gather_copy<<<(NB * IMG0 + 255) / 256, 256, 0, stream>>>(
        train_x, actT[0], (long)t * IMG0, (long)NT * IMG0, IMG0, NB);

    for (int j = 0; j < 5; ++j) {
      const int Npix = kHO[j] * kHO[j];
      if (j < 4) {
        dim3 grid(Npix / 32, kOC[j] / 16, NB);
        conv_fwd_wmma<2><<<grid, 32, 0, stream>>>(
            actT[j], fwW[j], fwB[j], actT[j + 1],
            kIC[j], kOC[j], kHI[j], kHI[j], kHO[j], kHO[j], kWS[j], kST[j],
            K[j], Kp[j], 1, 0, (long)WP[j], (long)kOC[j]);
      } else {
        dim3 grid(Npix / 16, kOC[j] / 16, NB);
        conv_fwd_wmma<1><<<grid, 32, 0, stream>>>(
            actT[j], fwW[j], fwB[j], actT[j + 1],
            kIC[j], kOC[j], kHI[j], kHI[j], kHO[j], kHO[j], kWS[j], kST[j],
            K[j], Kp[j], 1, 0, (long)WP[j], (long)kOC[j]);
      }
    }
    gate_mul<<<(NB * FEATK + 255) / 256, 256, 0, stream>>>(
        actT[5], featT, train_g, (long)t * FEATK, (long)NT * FEATK,
        (long)16 * FEATK, FEATK, NB);

    head_fwd_wmma<<<dim3(NCLSP / 32, NB), 32, 0, stream>>>(
        featT, fwWo, fwBo, logitT, /*taskMul=*/1, /*taskBase=*/0,
        (long)16 * FEATK, (long)16 * NCLSP);
    ce_train<<<NB, 128, 0, stream>>>(logitT, (long)16 * NCLSP, train_y, t, dlogitT);

    head_bwd_df<<<dim3(FEATK / 128, NB), 128, 0, stream>>>(
        dlogitT, fwWo, train_g, (long)t * FEATK, (long)NT * FEATK, dact5);
    head_upd<<<dim3((NCLSK * FEATK + 255) / 256, NB), 256, 0, stream>>>(
        fwWo, fwBo, dlogitT, featT, (long)16 * FEATK, log_lr);

    relu_mask<<<(NB * FEATK + 255) / 256, 256, 0, stream>>>(dact5, actT[5], NB * FEATK);
    float* gcur = dact5;
    float* gpp[2] = {g0, g1};
    int pp = 0;
    for (int j = 4; j >= 0; --j) {
      if (j > 0) {
        float* gnext = gpp[pp]; pp ^= 1;
        const int nIn = kIC[j] * kHI[j] * kHI[j];
        conv_bwd_x<<<dim3((nIn + 255) / 256, NB), 256, 0, stream>>>(
            gcur, fwW[j], gnext, kIC[j], kOC[j], kHI[j], kHI[j], kHO[j], kHO[j],
            kST[j], Kp[j], (long)WP[j]);
        conv_bwd_w_upd<<<dim3(kOC[j], NB), 256, 0, stream>>>(
            gcur, actT[j], fwW[j], fwB[j], log_lr,
            kIC[j], kOC[j], kHI[j], kHI[j], kHO[j], kHO[j], kST[j],
            Kp[j], (long)WP[j], (long)kOC[j]);
        relu_mask<<<(NB * nIn + 255) / 256, 256, 0, stream>>>(gnext, actT[j], NB * nIn);
        gcur = gnext;
      } else {
        conv_bwd_w_upd<<<dim3(kOC[0], NB), 256, 0, stream>>>(
            gcur, actT[0], fwW[0], fwB[0], log_lr,
            kIC[0], kOC[0], kHI[0], kHI[0], kHO[0], kHO[0], kST[0],
            Kp[0], (long)WP[0], (long)kOC[0]);
      }
    }
  }

  // ---- test pass: per task, 16 images ----
  for (int b = 0; b < NB; ++b) {
    gather_copy<<<(NL * IMG0 + 255) / 256, 256, 0, stream>>>(
        test_x, tpA, (long)b * NL * IMG0, (long)IMG0, IMG0, NL);
    float* src = tpA; float* dst = tpB;
    for (int j = 0; j < 5; ++j) {
      const int Npix = kHO[j] * kHO[j];
      if (j < 4) {
        dim3 grid(Npix / 32, kOC[j] / 16, NL);
        conv_fwd_wmma<2><<<grid, 32, 0, stream>>>(
            src, fwW[j], fwB[j], dst,
            kIC[j], kOC[j], kHI[j], kHI[j], kHO[j], kHO[j], kWS[j], kST[j],
            K[j], Kp[j], 0, b, (long)WP[j], (long)kOC[j]);
      } else {
        dim3 grid(Npix / 16, kOC[j] / 16, NL);
        conv_fwd_wmma<1><<<grid, 32, 0, stream>>>(
            src, fwW[j], fwB[j], dst,
            kIC[j], kOC[j], kHI[j], kHI[j], kHO[j], kHO[j], kWS[j], kST[j],
            K[j], Kp[j], 0, b, (long)WP[j], (long)kOC[j]);
      }
      float* tmp = src; src = dst; dst = tmp;
    }
    gate_mul<<<(NL * FEATK + 255) / 256, 256, 0, stream>>>(
        src, featE, test_g, (long)b * NL * FEATK, (long)FEATK, (long)FEATK, FEATK, NL);
    head_fwd_wmma<<<dim3(NCLSP / 32, 1), 32, 0, stream>>>(
        featE, fwWo, fwBo, logitE, /*taskMul=*/0, /*taskBase=*/b,
        0L, 0L);
    ce_test<<<NL, 128, 0, stream>>>(logitE, test_y, evalflg, b, out);
  }

  write_lr<<<1, 32, 0, stream>>>(log_lr, out + 128);
}
